// ABSA_Lstm_377957122440
// MI455X (gfx1250) — matrix operand order repaired and checked by reading, not compiled, not run on hardware
//
#include <hip/hip_runtime.h>
#include <hip/hip_bf16.h>
#include <math.h>

// ---------------- problem constants ----------------
#define B_   256      // batch
#define T_   128      // timesteps
#define H_   300      // hidden
#define D_   300      // embed dim
#define G4   1200     // 4*H
#define KP   320      // K padded to multiple of 32 (zero-filled)
#define C_   3        // classes

// ---------------- persistent-kernel shape ----------------
#define NBLK 64
#define TPB  256
#define NWAVE (NBLK * (TPB / 32))   // 512 waves for 1200 output tiles

typedef __attribute__((ext_vector_type(16))) __bf16         v16bf;
typedef __attribute__((ext_vector_type(8)))  float          v8f;
typedef __attribute__((ext_vector_type(8)))  unsigned short v8us;
typedef __attribute__((ext_vector_type(16))) unsigned short v16us;

// float -> bf16 (round-to-nearest-even), kept as raw u16 bits
__device__ __forceinline__ unsigned short f2bf(float f) {
  unsigned u = __float_as_uint(f);
  u += 0x7FFFu + ((u >> 16) & 1u);
  return (unsigned short)(u >> 16);
}

// Build a 16-element bf16 fragment from two contiguous 16B chunks.
__device__ __forceinline__ v16bf load_frag(const unsigned short* p0,
                                           const unsigned short* p1) {
  const v8us lo = *(const v8us*)p0;
  const v8us hi = *(const v8us*)p1;
  v16us r;
#pragma unroll
  for (int i = 0; i < 8; ++i) { r[i] = lo[i]; r[i + 8] = hi[i]; }
  return __builtin_bit_cast(v16bf, r);
}

// Device-wide split barrier (all NBLK blocks co-resident).
__device__ __forceinline__ void grid_sync(unsigned* cnt, unsigned* gen) {
  __syncthreads();
  if (threadIdx.x == 0) {
    __threadfence();
    unsigned g = __hip_atomic_load(gen, __ATOMIC_RELAXED, __HIP_MEMORY_SCOPE_AGENT);
    unsigned a = __hip_atomic_fetch_add(cnt, 1u, __ATOMIC_ACQ_REL, __HIP_MEMORY_SCOPE_AGENT);
    if (a == NBLK - 1) {
      __hip_atomic_store(cnt, 0u, __ATOMIC_RELAXED, __HIP_MEMORY_SCOPE_AGENT);
      __hip_atomic_fetch_add(gen, 1u, __ATOMIC_ACQ_REL, __HIP_MEMORY_SCOPE_AGENT);
    } else {
      while (__hip_atomic_load(gen, __ATOMIC_ACQUIRE, __HIP_MEMORY_SCOPE_AGENT) == g)
        __builtin_amdgcn_s_sleep(2);
    }
    __threadfence();
  }
  __syncthreads();
}

// ---------------- prep kernels ----------------
// Weights (4H x D) -> bf16 (4H x KP), zero-padded K.
__global__ void cvt_weight(const float* __restrict__ W, unsigned short* __restrict__ Wb) {
  const int r = blockIdx.x;   // 0..1199
  const int k = threadIdx.x;  // 0..319
  Wb[(size_t)r * KP + k] = (k < D_) ? f2bf(W[(size_t)r * D_ + k]) : (unsigned short)0;
}

// Embedding gather: XB[t*B + b][0..KP) = bf16(emb[sent[b][t]][:]) padded.
__global__ void gather_x(const int* __restrict__ sent, const float* __restrict__ emb,
                         unsigned short* __restrict__ XB) {
  const int row = blockIdx.x;      // t*B + b
  const int k   = threadIdx.x;     // 0..319
  const int t   = row >> 8;        // B_ = 256
  const int b   = row & (B_ - 1);
  const int w   = sent[b * T_ + t];
  XB[(size_t)row * KP + k] = (k < D_) ? f2bf(emb[(size_t)w * D_ + k]) : (unsigned short)0;
}

// State init: h/c fp32, h bf16 (padded), barrier counters.
__global__ void init_state(const float* __restrict__ h0, const float* __restrict__ c0,
                           unsigned short* __restrict__ HB, float* __restrict__ Hf,
                           float* __restrict__ Cf, unsigned* __restrict__ sync) {
  const int b = blockIdx.x;   // 0..255
  const int k = threadIdx.x;  // 0..319
  float hv = (k < H_) ? h0[(size_t)b * H_ + k] : 0.f;
  HB[(size_t)b * KP + k] = (k < H_) ? f2bf(hv) : (unsigned short)0;
  if (k < H_) {
    Hf[(size_t)b * H_ + k] = hv;
    Cf[(size_t)b * H_ + k] = c0[(size_t)b * H_ + k];
  }
  if (b == 0 && k < 2) sync[k] = 0u;
}

// ---------------- persistent recurrent kernel ----------------
__global__ void __launch_bounds__(TPB, 1)
lstm_recurrent(const int* __restrict__ lens,
               const float* __restrict__ b_ih, const float* __restrict__ b_hh,
               const float* __restrict__ Wout, const float* __restrict__ bout,
               const unsigned short* __restrict__ XB,
               const unsigned short* __restrict__ WihB,
               const unsigned short* __restrict__ WhhB,
               unsigned short* __restrict__ HB,
               float* __restrict__ Hf, float* __restrict__ Cf,
               float* __restrict__ G,
               unsigned* syncCnt, unsigned* syncGen,
               float* __restrict__ out) {
  const int tid  = threadIdx.x;
  const int lane = tid & 31;
  const int wave = blockIdx.x * (TPB / 32) + (tid >> 5);
  const int gtid = blockIdx.x * TPB + tid;
  const int mn   = lane & 15;          // M (A) / N (B,C,D) index within tile
  const int aoff = (lane >> 4) << 3;   // A-frag K sub-offset (0/8); also C/D M sub-offset
  const int boff = (lane >> 4) << 4;   // unused (0/16); B-frag K sub-offset is aoff

  for (int t = 0; t < T_; ++t) {
    const unsigned short* Xt = XB + (size_t)t * B_ * KP;

    // ---- Phase A: G = Xt*Wih^T + h*Whh^T (16x16 tiles, 16 x 75 grid) ----
    for (int tile = wave; tile < (B_ / 16) * (G4 / 16); tile += NWAVE) {
      const int tm = tile & 15;
      const int tn = tile >> 4;
      const unsigned short* arow0 = Xt   + (size_t)(tm * 16 + mn) * KP;
      const unsigned short* arow1 = HB   + (size_t)(tm * 16 + mn) * KP;
      const unsigned short* brow0 = WihB + (size_t)(tn * 16 + mn) * KP;
      const unsigned short* brow1 = WhhB + (size_t)(tn * 16 + mn) * KP;
      v8f acc = {0.f, 0.f, 0.f, 0.f, 0.f, 0.f, 0.f, 0.f};
#pragma unroll 2
      for (int kt = 0; kt < KP; kt += 32) {
        v16bf a = load_frag(arow0 + kt + aoff, arow0 + kt + aoff + 16);
        v16bf b = load_frag(brow0 + kt + aoff, brow0 + kt + aoff + 16);
        acc = __builtin_amdgcn_wmma_f32_16x16x32_bf16(false, a, false, b,
                                                      (short)0, acc, false, false);
      }
#pragma unroll 2
      for (int kt = 0; kt < KP; kt += 32) {
        v16bf a = load_frag(arow1 + kt + aoff, arow1 + kt + aoff + 16);
        v16bf b = load_frag(brow1 + kt + aoff, brow1 + kt + aoff + 16);
        acc = __builtin_amdgcn_wmma_f32_16x16x32_bf16(false, a, false, b,
                                                      (short)0, acc, false, false);
      }
      // C/D layout: VGPR r -> M = tm*16 + aoff + r, N = tn*16 + mn
      float* gdst = G + (size_t)(tm * 16 + aoff) * G4 + tn * 16 + mn;
#pragma unroll
      for (int r = 0; r < 8; ++r) gdst[(size_t)r * G4] = acc[r];
    }
    grid_sync(syncCnt, syncGen);

    // ---- Phase B: fused gate pointwise + state update + mask ----
    for (int e = gtid; e < B_ * H_; e += NBLK * TPB) {
      const int b = e / H_;
      const int j = e - b * H_;
      const float* gr = G + (size_t)b * G4;
      float gi = gr[j]          + b_ih[j]          + b_hh[j];
      float gf = gr[H_ + j]     + b_ih[H_ + j]     + b_hh[H_ + j];
      float gg = gr[2 * H_ + j] + b_ih[2 * H_ + j] + b_hh[2 * H_ + j];
      float go = gr[3 * H_ + j] + b_ih[3 * H_ + j] + b_hh[3 * H_ + j];
      float i_ = 1.0f / (1.0f + __expf(-gi));
      float f_ = 1.0f / (1.0f + __expf(-gf));
      float o_ = 1.0f / (1.0f + __expf(-go));
      float g_ = tanhf(gg);
      float cold = Cf[e];
      float hold = Hf[e];
      float cn = f_ * cold + i_ * g_;
      float hn = o_ * tanhf(cn);
      const bool live = (t < lens[b]);
      float cw = live ? cn : cold;
      float hw = live ? hn : hold;
      Cf[e] = cw;
      Hf[e] = hw;
      HB[(size_t)b * KP + j] = f2bf(hw);   // padded tail stays zero from init
    }
    grid_sync(syncCnt, syncGen);
  }

  // ---- Output head: out = h @ Wout^T + bout (256 x 3) ----
  for (int e = gtid; e < B_ * C_; e += NBLK * TPB) {
    const int b  = e / C_;
    const int cc = e - b * C_;
    const float* hrow = Hf + (size_t)b * H_;
    const float* wrow = Wout + (size_t)cc * H_;
    float s = bout[cc];
    for (int j = 0; j < H_; ++j) s += hrow[j] * wrow[j];
    out[e] = s;
  }
}

// ---------------- host entry ----------------
extern "C" void kernel_launch(void* const* d_in, const int* in_sizes, int n_in,
                              void* d_out, int out_size, void* d_ws, size_t ws_size,
                              hipStream_t stream) {
  (void)in_sizes; (void)n_in; (void)out_size; (void)ws_size;
  const int*   sent = (const int*)d_in[0];
  // d_in[1] = target (unused by the reference forward pass)
  const int*   lens = (const int*)d_in[2];
  const float* emb  = (const float*)d_in[3];
  const float* Wih  = (const float*)d_in[4];
  const float* Whh  = (const float*)d_in[5];
  const float* b_ih = (const float*)d_in[6];
  const float* b_hh = (const float*)d_in[7];
  const float* h0   = (const float*)d_in[8];
  const float* c0   = (const float*)d_in[9];
  const float* Wout = (const float*)d_in[10];
  const float* bout = (const float*)d_in[11];
  float* out = (float*)d_out;

  // workspace carve-out (~24.5 MB total), 256B aligned slices
  char* ws = (char*)d_ws;
  size_t off = 0;
  auto carve = [&](size_t bytes) -> char* {
    char* p = ws + off;
    off += (bytes + 255) & ~(size_t)255;
    return p;
  };
  unsigned short* WihB = (unsigned short*)carve((size_t)G4 * KP * 2);       // 0.77 MB
  unsigned short* WhhB = (unsigned short*)carve((size_t)G4 * KP * 2);       // 0.77 MB
  unsigned short* XB   = (unsigned short*)carve((size_t)B_ * T_ * KP * 2);  // 21.0 MB
  unsigned short* HB   = (unsigned short*)carve((size_t)B_ * KP * 2);       // 0.16 MB
  float* Hf = (float*)carve((size_t)B_ * H_ * 4);                           // 0.31 MB
  float* Cf = (float*)carve((size_t)B_ * H_ * 4);                           // 0.31 MB
  float* G  = (float*)carve((size_t)B_ * G4 * 4);                           // 1.23 MB
  unsigned* sync = (unsigned*)carve(256);

  cvt_weight<<<G4, KP, 0, stream>>>(Wih, WihB);
  cvt_weight<<<G4, KP, 0, stream>>>(Whh, WhhB);
  gather_x<<<B_ * T_, KP, 0, stream>>>(sent, emb, XB);
  init_state<<<B_, KP, 0, stream>>>(h0, c0, HB, Hf, Cf, sync);
  lstm_recurrent<<<NBLK, TPB, 0, stream>>>(lens, b_ih, b_hh, Wout, bout,
                                           XB, WihB, WhhB, HB, Hf, Cf, G,
                                           sync, sync + 1, out);
}
